// LSTMCell_11115375362476
// MI455X (gfx1250) — compile-verified, matrix operand order checked by reading, not run on hardware
//
#include <hip/hip_runtime.h>
#include <math.h>

// ---------------------------------------------------------------------------
// LSTM cell, fused:  gates = [x|h] @ [Wx;Wh]^ + (bx+bh)  -> elementwise
// B=16384, IN=512, H=512, K=IN+H=1024, N=4H=2048
// Strategy: split-bf16 WMMA GEMM (a_hi*b_hi + a_hi*b_lo + a_lo*b_hi),
// weights pre-transposed+split into d_ws (8MB, L2-resident), epilogue fused.
// v3: 512-thread blocks (16 waves, 4x4 wave grid), M=128 rows per block ->
//     halves L2 weight-fragment traffic vs the 64-row blocks, same per-wave
//     register footprint and inner loop.
// ---------------------------------------------------------------------------

#define B_   16384
#define IN_  512
#define H_   512
#define K_   1024
#define N_   2048

typedef __attribute__((ext_vector_type(16))) __bf16 bf16x16;
typedef __attribute__((ext_vector_type(8)))  __bf16 bf16x8;
typedef __attribute__((ext_vector_type(8)))  float  f32x8;
typedef __attribute__((ext_vector_type(4)))  float  f32x4;

// ---------------------------------------------------------------------------
// Prologue: transpose weights to [N, K], split each f32 into bf16 hi + lo.
// Also fold biases: bsum[n] = bx[n] + bh[n].
// threads: N_*K_ total, k fastest (idx = n*K_ + k) -> coalesced writes.
// ---------------------------------------------------------------------------
__global__ __launch_bounds__(256) void pack_weights(
    const float* __restrict__ Wx, const float* __restrict__ Wh,
    const float* __restrict__ bx, const float* __restrict__ bh,
    __bf16* __restrict__ wt_hi, __bf16* __restrict__ wt_lo,
    float* __restrict__ bsum)
{
    int idx = blockIdx.x * blockDim.x + threadIdx.x;   // n*K_ + k
    int k = idx & (K_ - 1);
    int n = idx >> 10;
    float w = (k < IN_) ? Wx[(size_t)k * N_ + n]
                        : Wh[(size_t)(k - IN_) * N_ + n];
    __bf16 hi = (__bf16)w;
    __bf16 lo = (__bf16)(w - (float)hi);
    wt_hi[idx] = hi;
    wt_lo[idx] = lo;
    if (k == 0) bsum[n] = bx[n] + bh[n];
}

// ---------------------------------------------------------------------------
// Main kernel: 16 waves/block (512 thr). Block computes 128 rows x 64 cols
// (per gate). Wave (wm 0..3, wn 0..3): 32 rows (2 WMMA M-tiles), 16 cols,
// all 4 gates -> elementwise epilogue is fully local to the wave.
// ---------------------------------------------------------------------------
__global__ __launch_bounds__(512) void lstm_wmma(
    const float* __restrict__ x, const float* __restrict__ h,
    const float* __restrict__ c,
    const __bf16* __restrict__ wt_hi, const __bf16* __restrict__ wt_lo,
    const float* __restrict__ bsum,
    float* __restrict__ h_new, float* __restrict__ c_new)
{
    const int lane = threadIdx.x & 31;
    const int wave = threadIdx.x >> 5;      // 0..15
    const int wm = wave >> 2;               // 0..3  (32-row group)
    const int wn = wave & 3;                // 0..3  (16-col group)
    const int r0 = blockIdx.x * 128 + wm * 32;
    const int c0 = blockIdx.y * 64 + wn * 16;

    const int lm = lane & 15;               // row (A) / col (B,D) within tile
    const int ko = (lane >> 4) * 8;         // per-lane K sub-offset (ISA layout)

    f32x8 acc[2][4] = {};                   // [mtile][gate]

    // Per-gate weight row pointers (transposed layout [N, K]); +ko baked in.
    const __bf16* bhp[4];
    const __bf16* blp[4];
#pragma unroll
    for (int g = 0; g < 4; ++g) {
        const size_t n = (size_t)(c0 + lm + g * H_);
        bhp[g] = wt_hi + n * K_ + ko;
        blp[g] = wt_lo + n * K_ + ko;
    }

    for (int kb = 0; kb < K_; kb += 32) {
        // ---- A fragments: load f32, split to bf16 hi/lo in registers ----
        const float* src  = (kb < IN_) ? x : h;
        const int    koff = (kb < IN_) ? kb : (kb - IN_);
        bf16x16 a_hi[2], a_lo[2];
#pragma unroll
        for (int mt = 0; mt < 2; ++mt) {
            const float* ap = src + (size_t)(r0 + mt * 16 + lm) * IN_ + koff + ko;
            f32x4 q0 = *(const f32x4*)(ap);        // K: ko+0..3
            f32x4 q1 = *(const f32x4*)(ap + 4);    // K: ko+4..7
            f32x4 q2 = *(const f32x4*)(ap + 16);   // K: ko+16..19
            f32x4 q3 = *(const f32x4*)(ap + 20);   // K: ko+20..23
#pragma unroll
            for (int e = 0; e < 4; ++e) {
                float v0 = q0[e], v1 = q1[e], v2 = q2[e], v3 = q3[e];
                __bf16 h0 = (__bf16)v0, h1 = (__bf16)v1;
                __bf16 h2 = (__bf16)v2, h3 = (__bf16)v3;
                a_hi[mt][e]      = h0; a_lo[mt][e]      = (__bf16)(v0 - (float)h0);
                a_hi[mt][e + 4]  = h1; a_lo[mt][e + 4]  = (__bf16)(v1 - (float)h1);
                a_hi[mt][e + 8]  = h2; a_lo[mt][e + 8]  = (__bf16)(v2 - (float)h2);
                a_hi[mt][e + 12] = h3; a_lo[mt][e + 12] = (__bf16)(v3 - (float)h3);
            }
        }

        // ---- B fragments per gate + 3-term split WMMA ----
#pragma unroll
        for (int g = 0; g < 4; ++g) {
            bf16x8 t0 = *(const bf16x8*)(bhp[g] + kb);
            bf16x8 t1 = *(const bf16x8*)(bhp[g] + kb + 16);
            bf16x8 t2 = *(const bf16x8*)(blp[g] + kb);
            bf16x8 t3 = *(const bf16x8*)(blp[g] + kb + 16);
            bf16x16 bHi, bLo;
#pragma unroll
            for (int e = 0; e < 8; ++e) {
                bHi[e] = t0[e]; bHi[e + 8] = t1[e];
                bLo[e] = t2[e]; bLo[e + 8] = t3[e];
            }
#pragma unroll
            for (int mt = 0; mt < 2; ++mt) {
                acc[mt][g] = __builtin_amdgcn_wmma_f32_16x16x32_bf16(
                    false, a_hi[mt], false, bHi, (short)0, acc[mt][g], false, false);
                acc[mt][g] = __builtin_amdgcn_wmma_f32_16x16x32_bf16(
                    false, a_hi[mt], false, bLo, (short)0, acc[mt][g], false, false);
                acc[mt][g] = __builtin_amdgcn_wmma_f32_16x16x32_bf16(
                    false, a_lo[mt], false, bHi, (short)0, acc[mt][g], false, false);
            }
        }
    }

    // ---- Fused LSTM epilogue (gates never hit global memory) ----
    float bias[4];
#pragma unroll
    for (int g = 0; g < 4; ++g) bias[g] = bsum[c0 + lm + g * H_];
    const int col = c0 + lm;

#pragma unroll
    for (int mt = 0; mt < 2; ++mt) {
#pragma unroll
        for (int e = 0; e < 8; ++e) {
            const int row = r0 + mt * 16 + e + 8 * (lane >> 4);   // D-matrix layout
            float gi = acc[mt][0][e] + bias[0];
            float gf = acc[mt][1][e] + bias[1];
            float gg = acc[mt][2][e] + bias[2];
            float go = acc[mt][3][e] + bias[3];
            float ig = 1.0f / (1.0f + __expf(-gi));
            float fg = 1.0f / (1.0f + __expf(-gf));
            float gt = tanhf(gg);
            float og = 1.0f / (1.0f + __expf(-go));
            const size_t off = (size_t)row * H_ + col;
            float cn = fg * c[off] + ig * gt;
            float hn = og * tanhf(cn);
            h_new[off] = hn;
            c_new[off] = cn;
        }
    }
}

// ---------------------------------------------------------------------------
extern "C" void kernel_launch(void* const* d_in, const int* in_sizes, int n_in,
                              void* d_out, int out_size, void* d_ws, size_t ws_size,
                              hipStream_t stream) {
    (void)in_sizes; (void)n_in; (void)out_size; (void)ws_size;
    const float* x  = (const float*)d_in[0];
    const float* h  = (const float*)d_in[1];
    const float* c  = (const float*)d_in[2];
    const float* Wx = (const float*)d_in[3];
    const float* Wh = (const float*)d_in[4];
    const float* bx = (const float*)d_in[5];
    const float* bh = (const float*)d_in[6];

    __bf16* wt_hi = (__bf16*)d_ws;                       // 2048*1024 bf16 = 4MB
    __bf16* wt_lo = wt_hi + (size_t)N_ * K_;             // 4MB
    float*  bsum  = (float*)(wt_lo + (size_t)N_ * K_);   // 8KB

    float* h_new = (float*)d_out;
    float* c_new = h_new + (size_t)B_ * H_;

    pack_weights<<<(N_ * K_) / 256, 256, 0, stream>>>(Wx, Wh, bx, bh,
                                                      wt_hi, wt_lo, bsum);
    dim3 grid(B_ / 128, H_ / 64);   // 128 x 8 blocks, 16 waves each
    lstm_wmma<<<grid, 512, 0, stream>>>(x, h, c, wt_hi, wt_lo, bsum,
                                        h_new, c_new);
}
